// allmodel_23776938950935
// MI455X (gfx1250) — matrix operand order repaired and checked by reading, not compiled, not run on hardware
//
#include <hip/hip_runtime.h>
#include <cstdint>
#include <cstddef>

// ---------------------------------------------------------------------------
// Types for WMMA
// ---------------------------------------------------------------------------
typedef _Float16 half_t;
typedef __attribute__((ext_vector_type(16))) _Float16 v16h;
typedef __attribute__((ext_vector_type(8)))  _Float16 v8h;
typedef __attribute__((ext_vector_type(8)))  float    v8f;

union HV16 { v16h v; v8h h[2]; };

// ---------------------------------------------------------------------------
// Prep: fp32 NHWC image [256,256,3] -> f16 padded [256*256, 32] (ch 3..31 = 0)
// ---------------------------------------------------------------------------
__global__ void prep_image_kernel(const float* __restrict__ img,
                                  half_t* __restrict__ act, int total)
{
    int idx = blockIdx.x * blockDim.x + threadIdx.x;
    if (idx >= total) return;
    int c = idx & 31;
    int p = idx >> 5;
    act[idx] = (c < 3) ? (half_t)img[p * 3 + c] : (half_t)0.0f;
}

// ---------------------------------------------------------------------------
// Prep: fp32 HWIO weights [3,3,Cin,Cout] -> f16 B-operand layout
//   wB[((ntile*KC + kc)*32 + lane)*16 + e]
// kc enumerates (ky,kx, cblock-of-32) in loop order (ky major, then kx, then cb).
// B-matrix 16-bit 32x16 layout (contiguous K order used here, matching the A loads): lane = n + 16*hi,
// element e holds B[K = hi*16 + e][n] of the 32-deep K chunk.
// Zero-padded beyond actual Cin/Cout.
// ---------------------------------------------------------------------------
__global__ void prep_wB_kernel(const float* __restrict__ w, half_t* __restrict__ wB,
                               int Cin, int Cout, int Cpin, int KC, int total)
{
    int idx = blockIdx.x * blockDim.x + threadIdx.x;
    if (idx >= total) return;
    int e    = idx & 15;
    int lane = (idx >> 4) & 31;
    int nk   = idx >> 9;                 // ntile*KC + kc
    int kc    = nk % KC;
    int ntile = nk / KC;
    int n  = lane & 15;
    int hi = lane >> 4;
    int cb32 = Cpin >> 5;
    int kk = kc / cb32;                  // ky*3 + kx
    int cb = kc - kk * cb32;
    int cin  = cb * 32 + hi * 16 + e;
    int cout = ntile * 16 + n;
    float v = 0.0f;
    if (cin < Cin && cout < Cout)
        v = w[((size_t)kk * Cin + cin) * Cout + cout];
    wB[idx] = (half_t)v;
}

// ---------------------------------------------------------------------------
// Implicit-GEMM 3x3 VALID conv via V_WMMA_F32_16X16X32_F16.
//   One wave per block computes a (MT*16 pixels) x (NT*16 channels) output
//   tile; K = 9 * Cpin in chunks of 32.  A loads are shared across the NT
//   column tiles, B loads across the MT row tiles -> 2 b128 loads per WMMA.
//   All addressing uses 32-bit running element offsets from the global kernel
//   args (buffers < 2 GB) so loads lower to global_load_b128 in saddr+voffset
//   form with folded immediates: no 64-bit VALU address math in the hot loop,
//   and no WMMA->VALU WAR hazard NOPs from operand-register recycling.
// A-operand per-lane addressing (contiguous K order, matching wB packing):
//   lane = m + 16*hi holds row M=m; elements 0..15 = K chunk c0+hi*16+{0..15}
//   (two 16-byte loads).
// EPI: 0 = relu -> f16 padded activations   (outStride = Cpout)
//      1 = +bias -> f32 logits              (outStride = 160, store ch < Cout)
//      2 = relu -> f32                      (store ch < Cout)
//      3 = hard_sigmoid -> f32 scalar map   (store ch == 0)
// ---------------------------------------------------------------------------
template<int EPI, int NT, int MT>
__global__ __launch_bounds__(32)
void conv3x3_wmma_kernel(const half_t* __restrict__ act,
                         const half_t* __restrict__ wB,
                         const float*  __restrict__ bias,
                         void* __restrict__ outp,
                         int Win, int Wout, int np, int Cpin, int KC,
                         int CoutActual, int outStride)
{
    const int lane = threadIdx.x;
    const int m    = lane & 15;
    const int hi   = lane >> 4;
    const int tile = blockIdx.x;
    const int nt0  = blockIdx.y * NT;

    // Per M-tile base element offset into `act` for this lane's A row
    // (clamped; stores are masked).  32-bit offsets throughout.
    unsigned aOff[MT];
#pragma unroll
    for (int mt = 0; mt < MT; ++mt) {
        int pix = (tile * MT + mt) * 16 + m;
        if (pix >= np) pix = np - 1;
        const int oy = pix / Wout;
        const int ox = pix - oy * Wout;
        aOff[mt] = (unsigned)(oy * Win + ox) * (unsigned)Cpin + (unsigned)(hi * 16);
    }

    v8f acc[NT][MT];
#pragma unroll
    for (int t = 0; t < NT; ++t)
#pragma unroll
        for (int mt = 0; mt < MT; ++mt)
            acc[t][mt] = v8f{0.f, 0.f, 0.f, 0.f, 0.f, 0.f, 0.f, 0.f};

    const int cb32 = Cpin >> 5;
    const unsigned rowStep = (unsigned)(Win * Cpin);      // advance one ky
    const unsigned ntStep  = (unsigned)KC * 512u;         // one N-tile of B

    // Running offsets, advanced by constants only.
    unsigned wIdx = (unsigned)nt0 * ntStep + (unsigned)(lane * 16);
    unsigned aRow[MT];
#pragma unroll
    for (int mt = 0; mt < MT; ++mt) aRow[mt] = aOff[mt];

    for (int ky = 0; ky < 3; ++ky) {
        unsigned aK[MT];
#pragma unroll
        for (int mt = 0; mt < MT; ++mt) aK[mt] = aRow[mt];
        for (int kx = 0; kx < 3; ++kx) {
            for (int cb = 0; cb < cb32; ++cb) {
                HV16 a[MT];
#pragma unroll
                for (int mt = 0; mt < MT; ++mt) {
                    a[mt].h[0] = *(const v8h*)(act + aK[mt]);       // K: c0+hi*16+0..7
                    a[mt].h[1] = *(const v8h*)(act + aK[mt] + 8);   // K: c0+hi*16+8..15
                    aK[mt] += 32u;
                }
#pragma unroll
                for (int t = 0; t < NT; ++t) {
                    const unsigned wi = wIdx + (unsigned)t * ntStep;
                    HV16 b;
                    b.h[0] = *(const v8h*)(wB + wi);
                    b.h[1] = *(const v8h*)(wB + wi + 8);
#pragma unroll
                    for (int mt = 0; mt < MT; ++mt)
                        acc[t][mt] = __builtin_amdgcn_wmma_f32_16x16x32_f16(
                            false, a[mt].v, false, b.v, (short)0, acc[t][mt],
                            false, false);
                }
                wIdx += 512u;
            }
            // next kx: advance one input pixel = Cpin elements; the cb loop
            // already advanced aK by Cpin (cb32*32), so nothing to add.
        }
#pragma unroll
        for (int mt = 0; mt < MT; ++mt) aRow[mt] += rowStep;
    }

    // Epilogue. D layout: lane holds column n = lane&15; VGPR r holds row
    // M = hi*8 + r  ->  output pixel (tile*MT+mt)*16 + hi*8 + r.
#pragma unroll
    for (int t = 0; t < NT; ++t) {
        const int ncg = (nt0 + t) * 16 + m;
        const float bb = (ncg < CoutActual) ? bias[ncg] : 0.0f;
#pragma unroll
        for (int mt = 0; mt < MT; ++mt) {
            const int mbase = (tile * MT + mt) * 16 + hi * 8;
            if (EPI == 0) {
                half_t* out = (half_t*)outp;
#pragma unroll
                for (int r = 0; r < 8; ++r) {
                    int mp = mbase + r;
                    if (mp < np) {
                        float v = acc[t][mt][r] + bb;
                        out[(unsigned)mp * (unsigned)outStride + (unsigned)ncg] =
                            (half_t)(v > 0.f ? v : 0.f);
                    }
                }
            } else if (EPI == 1) {
                float* out = (float*)outp;
                if (ncg < CoutActual) {
#pragma unroll
                    for (int r = 0; r < 8; ++r) {
                        int mp = mbase + r;
                        if (mp < np)
                            out[(unsigned)mp * (unsigned)outStride + (unsigned)ncg] =
                                acc[t][mt][r] + bb;
                    }
                }
            } else if (EPI == 2) {
                float* out = (float*)outp;
                if (ncg < CoutActual) {
#pragma unroll
                    for (int r = 0; r < 8; ++r) {
                        int mp = mbase + r;
                        if (mp < np)
                            out[(unsigned)mp * (unsigned)outStride + (unsigned)ncg] =
                                fmaxf(acc[t][mt][r] + bb, 0.f);
                    }
                }
            } else {
                float* out = (float*)outp;
                if (ncg == 0) {
#pragma unroll
                    for (int r = 0; r < 8; ++r) {
                        int mp = mbase + r;
                        if (mp < np) {
                            float v = (acc[t][mt][r] + bb) * 0.2f + 0.5f;
                            out[(unsigned)mp] = fminf(fmaxf(v, 0.f), 1.f);
                        }
                    }
                }
            }
        }
    }
}

// ---------------------------------------------------------------------------
// Misc small kernels
// ---------------------------------------------------------------------------
__global__ void zero_f32_kernel(float* __restrict__ p, int n)
{
    int i = blockIdx.x * blockDim.x + threadIdx.x;
    if (i < n) p[i] = 0.0f;
}

// Per-pixel softmax over 156 logits; accumulate colsum(S), S^T @ n, sum(e*dw)
// into accum = { c[156], sTn[312], num[1] } using LDS partials + global atomics.
__global__ __launch_bounds__(256)
void reduce_heads_kernel(const float* __restrict__ logits,  // [np][160]
                         const float* __restrict__ nbuf,    // [np][2]
                         const float* __restrict__ ebuf,    // [np]
                         const float* __restrict__ dw,      // [np]
                         float* __restrict__ accum, int np)
{
    __shared__ float sc[156];
    __shared__ float sn[312];
    __shared__ float se[1];
    const int tid = threadIdx.x;
    for (int j = tid; j < 156; j += 256) sc[j] = 0.0f;
    for (int j = tid; j < 312; j += 256) sn[j] = 0.0f;
    if (tid == 0) se[0] = 0.0f;
    __syncthreads();

    const int p = blockIdx.x * 256 + tid;
    if (p < np) {
        const float* row = logits + (size_t)p * 160;
        float mx = -1e30f;
        for (int j = 0; j < 156; ++j) mx = fmaxf(mx, row[j]);
        float s = 0.0f;
        for (int j = 0; j < 156; ++j) s += __expf(row[j] - mx);
        const float inv = 1.0f / s;
        const float n0 = nbuf[(size_t)p * 2 + 0];
        const float n1 = nbuf[(size_t)p * 2 + 1];
        for (int j = 0; j < 156; ++j) {
            float pr = __expf(row[j] - mx) * inv;
            atomicAdd(&sc[j], pr);
            atomicAdd(&sn[2 * j + 0], pr * n0);
            atomicAdd(&sn[2 * j + 1], pr * n1);
        }
        atomicAdd(&se[0], ebuf[p] * dw[p]);
    }
    __syncthreads();
    for (int j = tid; j < 156; j += 256) atomicAdd(&accum[j], sc[j]);
    for (int j = tid; j < 312; j += 256) atomicAdd(&accum[156 + j], sn[j]);
    if (tid == 0) atomicAdd(&accum[468], se[0]);
}

// out = [ node_features 156*2 | new_adj 156*156 | num 1 ]
__global__ void finalize_kernel(const float* __restrict__ accum,
                                const float* __restrict__ dense_b,
                                float* __restrict__ out)
{
    int idx = blockIdx.x * blockDim.x + threadIdx.x;
    if (idx < 312) {
        out[idx] = accum[156 + idx];                 // S^T @ n
    } else if (idx < 312 + 24336) {
        int k = idx - 312;
        int i = k / 156, j = k - i * 156;
        out[idx] = accum[i] * accum[j];              // outer(c, c)
    } else if (idx == 312 + 24336) {
        out[idx] = tanhf(accum[468] + dense_b[0]);   // num_nodes
    }
}

// ---------------------------------------------------------------------------
// Host launcher
// ---------------------------------------------------------------------------
static inline int pad32(int c) { return (c + 31) & ~31; }

extern "C" void kernel_launch(void* const* d_in, const int* in_sizes, int n_in,
                              void* d_out, int out_size, void* d_ws, size_t ws_size,
                              hipStream_t stream)
{
    (void)in_sizes; (void)n_in; (void)out_size; (void)ws_size;

    static const int CH[17] = {5, 8, 12, 16, 16, 32, 32, 32,
                               64, 64, 64, 96, 96, 128, 128, 176, 176};

    // Inputs in setup_inputs() dict-insertion order.
    const float* img = (const float*)d_in[0];
    const float* conv_w[17];
    const float* conv_b[17];
    for (int i = 0; i < 17; ++i) {
        conv_w[i] = (const float*)d_in[1 + i];
        conv_b[i] = (const float*)d_in[18 + i];
    }
    const float* num_w   = (const float*)d_in[35];
    const float* num_b   = (const float*)d_in[36];
    const float* dense_w = (const float*)d_in[37];
    const float* dense_b = (const float*)d_in[38];
    const float* adj_w   = (const float*)d_in[39];
    const float* adj_b   = (const float*)d_in[40];
    const float* node_w  = (const float*)d_in[41];
    const float* node_b  = (const float*)d_in[42];

    // Workspace carve (256-B aligned).
    char* base = (char*)d_ws;
    size_t off = 0;
    auto take = [&](size_t bytes) -> void* {
        off = (off + 255) & ~(size_t)255;
        void* p = base + off;
        off += bytes;
        return p;
    };
    const size_t MAX_ACT = 9633792;                  // 224*224*192 halves (max)
    half_t* actA   = (half_t*)take(MAX_ACT * 2);
    half_t* actB   = (half_t*)take(MAX_ACT * 2);
    half_t* wBbuf  = (half_t*)take((size_t)331776 * 2);  // 12 ntiles * 54 KC * 512
    float*  logits = (float*)take((size_t)48400 * 160 * 4);
    float*  nbuf   = (float*)take((size_t)48400 * 2 * 4);
    float*  ebuf   = (float*)take((size_t)48400 * 4);
    float*  accum  = (float*)take((size_t)469 * 4);

    // 1) Image -> f16 padded activations.
    {
        int total = 256 * 256 * 32;
        prep_image_kernel<<<(total + 255) / 256, 256, 0, stream>>>(img, actA, total);
    }

    // 2) 17 relu conv layers, ping-ponging actA/actB.
    half_t* cur = actA;
    half_t* nxt = actB;
    int Hin = 256, Cin = 3;
    for (int i = 0; i < 17; ++i) {
        const int Cout  = CH[i];
        const int Cpin  = pad32(Cin);
        const int Cpout = pad32(Cout);
        const int Win   = Hin, Wout = Hin - 2;
        const int np    = Wout * Wout;
        const int KC    = 9 * (Cpin / 32);
        const int ntiles = Cpout / 16;               // always even

        int total = ntiles * KC * 512;
        prep_wB_kernel<<<(total + 255) / 256, 256, 0, stream>>>(
            conv_w[i], wBbuf, Cin, Cout, Cpin, KC, total);

        dim3 grid((np + 31) / 32, ntiles / 2);       // MT=2, NT=2
        conv3x3_wmma_kernel<0, 2, 2><<<grid, 32, 0, stream>>>(
            cur, wBbuf, conv_b[i], (void*)nxt, Win, Wout, np, Cpin, KC, Cout, Cpout);

        half_t* t = cur; cur = nxt; nxt = t;
        Hin = Wout; Cin = Cout;
    }
    // cur = org activations [222,222,192] f16 (176 real channels).

    const int Win = 222, Wout = 220, np = 48400, Cpin = 192, KC = 54;

    // 3a) Adj head: 176 -> 156 logits (pad 160), f32, no activation.
    {
        int total = 10 * KC * 512;
        prep_wB_kernel<<<(total + 255) / 256, 256, 0, stream>>>(
            adj_w, wBbuf, 176, 156, Cpin, KC, total);
        dim3 grid((np + 31) / 32, 5);                // MT=2, NT=2
        conv3x3_wmma_kernel<1, 2, 2><<<grid, 32, 0, stream>>>(
            cur, wBbuf, adj_b, (void*)logits, Win, Wout, np, Cpin, KC, 156, 160);
    }
    // 3b) Node head: 176 -> 2, relu, f32.
    {
        int total = 1 * KC * 512;
        prep_wB_kernel<<<(total + 255) / 256, 256, 0, stream>>>(
            node_w, wBbuf, 176, 2, Cpin, KC, total);
        dim3 grid((np + 31) / 32, 1);                // MT=2, NT=1
        conv3x3_wmma_kernel<2, 1, 2><<<grid, 32, 0, stream>>>(
            cur, wBbuf, node_b, (void*)nbuf, Win, Wout, np, Cpin, KC, 2, 2);
    }
    // 3c) Num head: 176 -> 1, hard_sigmoid, f32.
    {
        int total = 1 * KC * 512;
        prep_wB_kernel<<<(total + 255) / 256, 256, 0, stream>>>(
            num_w, wBbuf, 176, 1, Cpin, KC, total);
        dim3 grid((np + 31) / 32, 1);                // MT=2, NT=1
        conv3x3_wmma_kernel<3, 1, 2><<<grid, 32, 0, stream>>>(
            cur, wBbuf, num_b, (void*)ebuf, Win, Wout, np, Cpin, KC, 1, 1);
    }

    // 4) Reductions.
    zero_f32_kernel<<<2, 256, 0, stream>>>(accum, 469);
    reduce_heads_kernel<<<(np + 255) / 256, 256, 0, stream>>>(
        logits, nbuf, ebuf, dense_w, accum, np);

    // 5) Assemble outputs.
    finalize_kernel<<<(24649 + 255) / 256, 256, 0, stream>>>(
        accum, dense_b, (float*)d_out);
}
